// GLMAttention_31129922962159
// MI455X (gfx1250) — matrix operand order repaired and checked by reading, not compiled, not run on hardware
//
#include <hip/hip_runtime.h>
#include <hip/hip_bf16.h>
#include <cstdint>

#define HID 4096
#define NH 32
#define NKV 2
#define HD 128
#define ROT 64
#define B_ 2
#define L_ 2048
#define QKV_OUT (HID + 2 * HD * NKV) /* 4608 */
#define SCPAD 2052                    /* 2048 + 4 pad: avoid LDS bank conflicts */
#define KSTEP 64

typedef __attribute__((ext_vector_type(16))) __bf16 v16bf;
typedef __attribute__((ext_vector_type(8)))  float  v8f;
typedef __attribute__((ext_vector_type(4)))  unsigned int v4u;
typedef __attribute__((ext_vector_type(8)))  int v8i;
typedef __attribute__((ext_vector_type(4)))  int v4i;

union Frag16 { uint4 u[2]; v16bf v; };
union Acc8   { v8f v; float f[8]; };

#define HAVE_TDM __has_builtin(__builtin_amdgcn_tensor_load_to_lds)

__device__ __forceinline__ unsigned short f32_bf16(float x) {
  unsigned u = __float_as_uint(x);
  u += 0x7fffu + ((u >> 16) & 1u);   // round-to-nearest-even
  return (unsigned short)(u >> 16);
}
__device__ __forceinline__ unsigned pack2(float lo, float hi) {
  return (unsigned)f32_bf16(lo) | ((unsigned)f32_bf16(hi) << 16);
}

#if HAVE_TDM
// Issue one TDM 2D tile load: global (bf16, row-major, row_stride elems)
// -> LDS packed tile [tile_rows][tile_k]. D# per CDNA5 ISA.
__device__ __forceinline__ void tdm_load_2d(unsigned lds_addr, const void* gaddr,
                                            unsigned tile_k, unsigned tile_rows,
                                            unsigned long long row_stride,
                                            unsigned tensor_d0, unsigned tensor_d1) {
  unsigned long long ga = (unsigned long long)(uintptr_t)gaddr;
  v4u g0 = { 1u,                                    // count=1, user mode
             lds_addr,                              // lds_addr (bytes)
             (unsigned)ga,                          // global_addr[31:0]
             (unsigned)((ga >> 32) & 0x01FFFFFFu) | (2u << 30) }; // addr[56:32] | type=2
  v8i g1 = { (int)(1u << 16),                                    // data_size=1 (2B)
             (int)((tensor_d0 & 0xFFFFu) << 16),                 // tensor_dim0[15:0]
             (int)((tensor_d0 >> 16) | ((tensor_d1 & 0xFFFFu) << 16)),
             (int)((tensor_d1 >> 16) | ((tile_k & 0xFFFFu) << 16)), // tile_dim0
             (int)(tile_rows & 0xFFFFu),                         // tile_dim1 (tile_dim2=0)
             (int)(unsigned)(row_stride & 0xFFFFFFFFull),        // dim0_stride[31:0]
             (int)((unsigned)(row_stride >> 32) & 0xFFFFu),      // dim0_stride[47:32]
             0 };
  v4i z4 = { 0, 0, 0, 0 };
#if __clang_major__ >= 23
  v8i z8 = { 0, 0, 0, 0, 0, 0, 0, 0 };
  __builtin_amdgcn_tensor_load_to_lds(g0, g1, z4, z4, z8, 0);
#else
  __builtin_amdgcn_tensor_load_to_lds(g0, g1, z4, z4, 0);
#endif
}
#endif

// ---------------------------------------------------------------- converts
__global__ void __launch_bounds__(256)
f32_to_bf16(const float* __restrict__ s, unsigned short* __restrict__ d, size_t n) {
  size_t i = (size_t)blockIdx.x * blockDim.x + threadIdx.x;
  size_t stride = (size_t)gridDim.x * blockDim.x;
  for (; i < n; i += stride) d[i] = f32_bf16(s[i]);
}

// ---------------------------------------------------------------- GEMM
// C[M,N] = A[M,K](bf16) * Wt[N,K]^T(bf16) + bias ; f32 accumulate via WMMA.
// Block: 256 thr = 8 waves as 2(M) x 4(N); wave tile 32x32; block tile 64x128.
// K staged in 64-wide tiles: TDM (async DMA) double-buffered into LDS.
__global__ void __launch_bounds__(256)
gemm_bf16_nt(const unsigned short* __restrict__ A,
             const unsigned short* __restrict__ Wt,
             const float* __restrict__ bias,
             float* __restrict__ C, int M, int N, int K) {
  __shared__ unsigned short sA[2][64 * KSTEP];   // 2 x 8KB
  __shared__ unsigned short sB[2][128 * KSTEP];  // 2 x 16KB

  const int lane = threadIdx.x & 31;
  const int wave = threadIdx.x >> 5;
  const int wm = wave & 1;
  const int wn = wave >> 1;
  const int lr = lane & 15;
  const int lh = lane >> 4;
  const int bm0 = blockIdx.y * 64;
  const int bn0 = blockIdx.x * 128;
  const size_t sK = (size_t)K;
  const int nkb = K / KSTEP;

  Acc8 acc[2][2];
#pragma unroll
  for (int i = 0; i < 2; i++)
#pragma unroll
    for (int j = 0; j < 2; j++)
#pragma unroll
      for (int r = 0; r < 8; r++) acc[i][j].f[r] = 0.f;

#if HAVE_TDM
  if (wave == 0) {
    tdm_load_2d((unsigned)(uintptr_t)&sA[0][0], A + (size_t)bm0 * sK, KSTEP, 64,
                (unsigned long long)K, (unsigned)K, (unsigned)M);
    tdm_load_2d((unsigned)(uintptr_t)&sB[0][0], Wt + (size_t)bn0 * sK, KSTEP, 128,
                (unsigned long long)K, (unsigned)K, (unsigned)N);
  }
#endif

  for (int kb = 0; kb < nkb; kb++) {
    const int buf = kb & 1;
#if HAVE_TDM
    if (wave == 0) {
      if (kb + 1 < nkb) {
        const size_t ko = (size_t)(kb + 1) * KSTEP;
        tdm_load_2d((unsigned)(uintptr_t)&sA[buf ^ 1][0], A + (size_t)bm0 * sK + ko,
                    KSTEP, 64, (unsigned long long)K, (unsigned)K, (unsigned)M);
        tdm_load_2d((unsigned)(uintptr_t)&sB[buf ^ 1][0], Wt + (size_t)bn0 * sK + ko,
                    KSTEP, 128, (unsigned long long)K, (unsigned)K, (unsigned)N);
        __builtin_amdgcn_s_wait_tensorcnt((short)2); // current stage done, next in flight
      } else {
        __builtin_amdgcn_s_wait_tensorcnt((short)0);
      }
    }
#else
    { // cooperative global -> LDS copy fallback
      const int tid = threadIdx.x;
      const uint4* ga = (const uint4*)(A + (size_t)bm0 * sK + (size_t)kb * KSTEP);
      const uint4* gb = (const uint4*)(Wt + (size_t)bn0 * sK + (size_t)kb * KSTEP);
      uint4* la = (uint4*)&sA[buf][0];
      uint4* lb = (uint4*)&sB[buf][0];
      const int kq = sK >> 3; // uint4 per global row
#pragma unroll
      for (int t = 0; t < 2; t++) { // 64 rows * 8 uint4 = 512
        int f = tid + t * 256, row = f >> 3, c = f & 7;
        la[f] = ga[(size_t)row * kq + c];
      }
#pragma unroll
      for (int t = 0; t < 4; t++) { // 128 rows * 8 uint4 = 1024
        int f = tid + t * 256, row = f >> 3, c = f & 7;
        lb[f] = gb[(size_t)row * kq + c];
      }
    }
#endif
    __syncthreads();

    const unsigned short* tA = &sA[buf][0];
    const unsigned short* tB = &sB[buf][0];
#pragma unroll
    for (int kk = 0; kk < KSTEP; kk += 32) {
      Frag16 a[2], b[2];
#pragma unroll
      for (int mi = 0; mi < 2; mi++) {
        const uint4* p = (const uint4*)(tA + (wm * 32 + mi * 16 + lr) * KSTEP + kk + lh * 8);
        a[mi].u[0] = p[0];          // K: kk + lh*8 .. +7
        a[mi].u[1] = p[2];          // K: kk + 16 + lh*8 .. +7
      }
#pragma unroll
      for (int ni = 0; ni < 2; ni++) {
        const uint4* p = (const uint4*)(tB + (wn * 32 + ni * 16 + lr) * KSTEP + kk + lh * 8);
        b[ni].u[0] = p[0];          // K: kk + lh*8 .. +7
        b[ni].u[1] = p[2];          // K: kk + 16 + lh*8 .. +7
      }
#pragma unroll
      for (int mi = 0; mi < 2; mi++)
#pragma unroll
        for (int ni = 0; ni < 2; ni++)
          acc[mi][ni].v = __builtin_amdgcn_wmma_f32_16x16x32_bf16(
              false, a[mi].v, false, b[ni].v, (short)0, acc[mi][ni].v, false, false);
    }
    __syncthreads();
  }

#pragma unroll
  for (int mi = 0; mi < 2; mi++)
#pragma unroll
    for (int ni = 0; ni < 2; ni++) {
      const int col = bn0 + wn * 32 + ni * 16 + lr;
      const float bv = bias ? bias[col] : 0.f;
#pragma unroll
      for (int r = 0; r < 8; r++) {
        const int row = bm0 + wm * 32 + mi * 16 + r + lh * 8;
        C[(size_t)row * N + col] = acc[mi][ni].f[r] + bv;
      }
    }
}

// ---------------------------------------------------------------- RoPE + pack
__global__ void __launch_bounds__(256)
rope_pack(const float* __restrict__ qkv,
          unsigned short* __restrict__ qB,
          unsigned short* __restrict__ kB,
          unsigned short* __restrict__ vT) {
  const int bl = blockIdx.x;
  const int b = bl / L_;
  const int l = bl % L_;
  const float* x = qkv + (size_t)bl * QKV_OUT;
  const float nlb = -__logf(10000.0f) / 32.0f;

  for (int o = threadIdx.x; o < QKV_OUT; o += 256) {
    float val = x[o];
    if (o < HID + HD * NKV) {            // q or k -> apply RoPE
      const int d = o & 127;
      float out = val;
      if (d < ROT) {
        const int i = d & 31;
        const float ang = (float)l * __expf((float)i * nlb);
        float s, c;
        __sincosf(ang, &s, &c);
        out = (d < 32) ? (val * c - x[o + 32] * s)
                       : (val * c + x[o - 32] * s);
      }
      if (o < HID) {
        const int h = o >> 7;
        qB[((size_t)(b * NH + h) * L_ + l) * HD + d] = f32_bf16(out);
      } else {
        const int kv = (o - HID) >> 7;
        kB[((size_t)(b * NKV + kv) * L_ + l) * HD + d] = f32_bf16(out);
      }
    } else {                             // v -> transpose store [HD][L]
      const int oo = o - HID - HD * NKV;
      const int kv = oo >> 7, d = oo & 127;
      vT[((size_t)(b * NKV + kv) * HD + d) * L_ + l] = f32_bf16(val);
    }
  }
}

// ---------------------------------------------------------------- attention
// One wave per (b, h, 16-query-row tile). Scores for the full causal key
// range live in LDS (up to 16 x 2048 f32 = 128KB; CDNA5 WGP has 320KB LDS).
__global__ void __launch_bounds__(32)
attention_fused(const unsigned short* __restrict__ qB,
                const unsigned short* __restrict__ kB,
                const unsigned short* __restrict__ vT,
                unsigned short* __restrict__ attnB) {
  extern __shared__ float sc[]; // [16][SCPAD]
  const int qt = blockIdx.x, h = blockIdx.y, b = blockIdx.z;
  const int kv = h / (NH / NKV);
  const int lane = threadIdx.x;
  const int lr = lane & 15, lh = lane >> 4;
  const int qbase = qt * 16;
  const int kend = qbase + 16;                 // causal limit (exclusive)
  const int kcov = ((kend + 31) >> 5) << 5;    // round up to 32-key tiles
  const float scale = 0.08838834764831845f;    // 1/sqrt(128)

  Frag16 aq[4];
  {
    const unsigned short* qrow = qB + ((size_t)(b * NH + h) * L_ + qbase + lr) * HD;
#pragma unroll
    for (int kc = 0; kc < 4; kc++) {
      const uint4* p = (const uint4*)(qrow + kc * 32 + lh * 8);
      aq[kc].u[0] = p[0];
      aq[kc].u[1] = p[2];
    }
  }

  const unsigned short* kbase = kB + (size_t)(b * NKV + kv) * L_ * HD;

  // ---- Phase 1: S = Q K^T, scaled + causal-masked, into LDS
  for (int kt = 0; kt < kcov; kt += 32) {
#pragma unroll
    for (int ni = 0; ni < 2; ni++) {
      Acc8 s;
#pragma unroll
      for (int r = 0; r < 8; r++) s.f[r] = 0.f;
#pragma unroll
      for (int kc = 0; kc < 4; kc++) {
        Frag16 bk;
        const uint4* p = (const uint4*)(kbase + (size_t)(kt + ni * 16 + lr) * HD + kc * 32 + lh * 8);
        bk.u[0] = p[0];
        bk.u[1] = p[2];
        s.v = __builtin_amdgcn_wmma_f32_16x16x32_bf16(
            false, aq[kc].v, false, bk.v, (short)0, s.v, false, false);
      }
      const int col = kt + ni * 16 + lr;
#pragma unroll
      for (int r = 0; r < 8; r++) {
        const int row = r + lh * 8;
        sc[row * SCPAD + col] = (col <= qbase + row) ? s.f[r] * scale : -1e30f;
      }
    }
  }
  __syncthreads();

  // ---- Phase 2: exact softmax; 2 lanes per row (parity lh), shfl_xor(16) join
  {
    float* rp = sc + lr * SCPAD;
    float mx = -1e30f;
    for (int j = lh; j < kcov; j += 2) mx = fmaxf(mx, rp[j]);
    mx = fmaxf(mx, __shfl_xor(mx, 16, 32));
    float sum = 0.f;
    for (int j = lh; j < kcov; j += 2) {
      float e = __expf(rp[j] - mx);
      rp[j] = e;
      sum += e;
    }
    sum += __shfl_xor(sum, 16, 32);
    const float inv = 1.f / sum;
    for (int j = lh; j < kcov; j += 2) rp[j] *= inv;
  }
  __syncthreads();

  // ---- Phase 3: O = P V ; P read from LDS (ds_load_b128) + f32->bf16 pack
  Acc8 o[8];
#pragma unroll
  for (int ni = 0; ni < 8; ni++)
#pragma unroll
    for (int r = 0; r < 8; r++) o[ni].f[r] = 0.f;

  const unsigned short* vbase = vT + (size_t)(b * NKV + kv) * HD * L_;
  for (int kt = 0; kt < kcov; kt += 32) {
    Frag16 ap;
    const float* rp = sc + lr * SCPAD + kt + lh * 8;
    float4 f0 = *(const float4*)(rp);
    float4 f1 = *(const float4*)(rp + 4);
    float4 f2 = *(const float4*)(rp + 16);
    float4 f3 = *(const float4*)(rp + 20);
    ap.u[0] = make_uint4(pack2(f0.x, f0.y), pack2(f0.z, f0.w),
                         pack2(f1.x, f1.y), pack2(f1.z, f1.w));
    ap.u[1] = make_uint4(pack2(f2.x, f2.y), pack2(f2.z, f2.w),
                         pack2(f3.x, f3.y), pack2(f3.z, f3.w));
#pragma unroll
    for (int ni = 0; ni < 8; ni++) {
      Frag16 bv;
      const uint4* p = (const uint4*)(vbase + (size_t)(ni * 16 + lr) * L_ + kt + lh * 8);
      bv.u[0] = p[0];
      bv.u[1] = p[2];
      o[ni].v = __builtin_amdgcn_wmma_f32_16x16x32_bf16(
          false, ap.v, false, bv.v, (short)0, o[ni].v, false, false);
    }
  }

#pragma unroll
  for (int ni = 0; ni < 8; ni++) {
    const int col = h * HD + ni * 16 + lr;
#pragma unroll
    for (int r = 0; r < 8; r++) {
      const int row = qbase + r + lh * 8;
      attnB[(size_t)(b * L_ + row) * HID + col] = f32_bf16(o[ni].f[r]);
    }
  }
}

// ---------------------------------------------------------------- launch
extern "C" void kernel_launch(void* const* d_in, const int* in_sizes, int n_in,
                              void* d_out, int out_size, void* d_ws, size_t ws_size,
                              hipStream_t stream) {
  (void)in_sizes; (void)n_in; (void)out_size; (void)ws_size;
  const float* hs      = (const float*)d_in[0]; // [B,L,HID]
  const float* qkv_w   = (const float*)d_in[1]; // [4608,HID]
  const float* qkv_b   = (const float*)d_in[2]; // [4608]
  const float* dense_w = (const float*)d_in[3]; // [HID,HID]

  char* ws = (char*)d_ws;
  auto alloc = [&](size_t bytes) -> char* {
    char* p = ws;
    ws += (bytes + 255) & ~(size_t)255;
    return p;
  };
  unsigned short* hsB   = (unsigned short*)alloc((size_t)B_ * L_ * HID * 2);
  unsigned short* wqkvB = (unsigned short*)alloc((size_t)QKV_OUT * HID * 2);
  unsigned short* wdB   = (unsigned short*)alloc((size_t)HID * HID * 2);
  float*          qkvF  = (float*)alloc((size_t)B_ * L_ * QKV_OUT * 4);
  unsigned short* qBuf  = (unsigned short*)alloc((size_t)B_ * NH * L_ * HD * 2);
  unsigned short* kBuf  = (unsigned short*)alloc((size_t)B_ * NKV * L_ * HD * 2);
  unsigned short* vTBuf = (unsigned short*)alloc((size_t)B_ * NKV * HD * L_ * 2);
  unsigned short* attnB = (unsigned short*)alloc((size_t)B_ * L_ * HID * 2);

  f32_to_bf16<<<2048, 256, 0, stream>>>(hs, hsB, (size_t)B_ * L_ * HID);
  f32_to_bf16<<<2048, 256, 0, stream>>>(qkv_w, wqkvB, (size_t)QKV_OUT * HID);
  f32_to_bf16<<<2048, 256, 0, stream>>>(dense_w, wdB, (size_t)HID * HID);

  gemm_bf16_nt<<<dim3(QKV_OUT / 128, (B_ * L_) / 64), 256, 0, stream>>>(
      hsB, wqkvB, qkv_b, qkvF, B_ * L_, QKV_OUT, HID);

  rope_pack<<<B_ * L_, 256, 0, stream>>>(qkvF, qBuf, kBuf, vTBuf);

  attention_fused<<<dim3(L_ / 16, NH, B_), 32, 16 * SCPAD * sizeof(float), stream>>>(
      qBuf, kBuf, vTBuf, attnB);

  gemm_bf16_nt<<<dim3(HID / 128, (B_ * L_) / 64), 256, 0, stream>>>(
      attnB, wdB, nullptr, (float*)d_out, B_ * L_, HID, HID);
}
